// EncoderDecoder_57294863728958
// MI455X (gfx1250) — matrix operand order repaired and checked by reading, not compiled, not run on hardware
//
#include <hip/hip_runtime.h>
#include <cstdint>

// ============================================================================
// MI455X (gfx1250) GRU encoder-decoder.
//   - All GEMMs via v_wmma_f32_16x16x32_f16 (f32 accumulate).
//   - Recurrent h staged to LDS each step via TDM tensor_load_to_lds with
//     descriptor padding (row stride 512+8 halves) + s_wait_tensorcnt.
//   - Persistent kernels, grid barrier per timestep (atomics, acq/rel).
//   - Weights pre-packed into exact WMMA B-fragment layout.
//   - launch_bounds(256,2): cap VGPRs so 8 waves stay resident per WGP.
// ============================================================================

typedef _Float16 half_t;
typedef __attribute__((ext_vector_type(16))) _Float16 v16h;
typedef __attribute__((ext_vector_type(8)))  float    v8f;
typedef __attribute__((ext_vector_type(4)))  float    f4;
typedef unsigned int u32;
typedef __attribute__((ext_vector_type(4))) unsigned int u32x4;
typedef __attribute__((ext_vector_type(8))) int i32x8;
typedef __attribute__((ext_vector_type(4))) int i32x4;

#define B_   128
#define T_   512
#define U_   256
#define H_   512
#define NG   1536        // 3*H
#define LDH  520         // LDS row stride in halves: 512 data + 8 pad

// ---------------------------------------------------------------------------
// WMMA + fragment helpers
// ---------------------------------------------------------------------------
static __device__ __forceinline__ v8f wmma16(v16h a, v16h b, v8f c) {
  return __builtin_amdgcn_wmma_f32_16x16x32_f16(false, a, false, b, (short)0, c,
                                                false, false);
}

union FragU { v16h h; f4 f[2]; };

static __device__ __forceinline__ v8f zero8() {
  v8f r;
#pragma unroll
  for (int i = 0; i < 8; i++) r[i] = 0.0f;
  return r;
}

// Packed B fragment: tile = 32 lanes x 16 halves contiguous (1KB).
static __device__ __forceinline__ v16h ld_b_frag(const half_t* packed, int ntile,
                                                 int ktile, int KT) {
  const int lane = threadIdx.x & 31;
  const half_t* p = packed + (((size_t)ntile * KT + ktile) * 32 + lane) * 16;
  FragU u;
  u.f[0] = *(const f4*)p;
  u.f[1] = *(const f4*)(p + 8);
  return u.h;
}

// A fragment from row-major f16 in LDS (row stride LDH halves).
static __device__ __forceinline__ v16h ld_a_lds(const half_t* base, int m0, int k0) {
  const int lane = threadIdx.x & 31;
  const half_t* p = base + (size_t)(m0 + (lane & 15)) * LDH + k0 + ((lane >> 4) << 3);
  FragU u;
  u.f[0] = *(const f4*)p;
  u.f[1] = *(const f4*)(p + 16);
  return u.h;
}

// A fragment from row-major f16 in global (p = &A[row][klo]).
static __device__ __forceinline__ v16h ld_a_f16g(const half_t* p) {
  FragU u;
  u.f[0] = *(const f4*)p;
  u.f[1] = *(const f4*)(p + 16);
  return u.h;
}

// A fragment from row-major f32 in global (converted to f16).
static __device__ __forceinline__ v16h ld_a_f32(const float* p) {
  f4 a = *(const f4*)(p), b = *(const f4*)(p + 4);
  f4 c = *(const f4*)(p + 16), d = *(const f4*)(p + 20);
  v16h r;
#pragma unroll
  for (int i = 0; i < 4; i++) {
    r[i] = (half_t)a[i]; r[4 + i] = (half_t)b[i];
    r[8 + i] = (half_t)c[i]; r[12 + i] = (half_t)d[i];
  }
  return r;
}

static __device__ __forceinline__ float sigm(float v) {
  return 1.0f / (1.0f + expf(-v));
}

// ---------------------------------------------------------------------------
// TDM: DMA a (rows x 512) f16 tile, global -> LDS, padding 8 halves per row
// (pad_interval=7 -> every 256 dwords = one row; pad_amount code 3 -> 4 dwords).
// ---------------------------------------------------------------------------
static __device__ __forceinline__ void tdm_load_tile(u32 lds_off_bytes,
                                                     const half_t* gsrc, int rows) {
  unsigned long long ga = (unsigned long long)(uintptr_t)gsrc;
  u32x4 g0;
  g0[0] = 1u;                                            // count=1
  g0[1] = lds_off_bytes;                                 // lds_addr
  g0[2] = (u32)ga;                                       // global_addr[31:0]
  g0[3] = (u32)((ga >> 32) & 0x1FFFFFFull) | (2u << 30); // addr[56:32] | type=2
  i32x8 g1;
  g1[0] = (1 << 16) | (1 << 20) | (7 << 22) | (3 << 25); // data_size=2B,pad on
  g1[1] = (int)(((u32)H_ & 0xFFFFu) << 16);              // tensor_dim0 lo16
  g1[2] = (int)(((u32)H_ >> 16) | (((u32)rows & 0xFFFFu) << 16)); // dim0 hi|dim1 lo
  g1[3] = (int)(((u32)rows >> 16) | (((u32)H_ & 0xFFFFu) << 16)); // dim1 hi|tile0
  g1[4] = (int)((u32)rows & 0xFFFFu);                    // tile_dim1, tile_dim2=0
  g1[5] = (int)H_;                                       // tensor_dim0_stride
  g1[6] = 0;
  g1[7] = 0;
  i32x4 z4 = {0, 0, 0, 0};
#if defined(__clang_major__) && (__clang_major__ >= 23)
  i32x8 z8 = {0, 0, 0, 0, 0, 0, 0, 0};
  __builtin_amdgcn_tensor_load_to_lds(g0, g1, z4, z4, z8, 0);
#else
  __builtin_amdgcn_tensor_load_to_lds(g0, g1, z4, z4, 0);
#endif
}

// ---------------------------------------------------------------------------
// Device-wide barrier (generation counter, acq/rel, s_sleep spin)
// ---------------------------------------------------------------------------
static __device__ __forceinline__ void grid_barrier(int* cnt, int* gen, int nwg) {
  __syncthreads();
  if (threadIdx.x == 0) {
    __threadfence();
    int g = __hip_atomic_load(gen, __ATOMIC_RELAXED, __HIP_MEMORY_SCOPE_AGENT);
    if (__hip_atomic_fetch_add(cnt, 1, __ATOMIC_ACQ_REL, __HIP_MEMORY_SCOPE_AGENT)
        == nwg - 1) {
      __hip_atomic_store(cnt, 0, __ATOMIC_RELAXED, __HIP_MEMORY_SCOPE_AGENT);
      __hip_atomic_fetch_add(gen, 1, __ATOMIC_RELEASE, __HIP_MEMORY_SCOPE_AGENT);
    } else {
      while (__hip_atomic_load(gen, __ATOMIC_ACQUIRE, __HIP_MEMORY_SCOPE_AGENT) == g)
        __builtin_amdgcn_s_sleep(2);
    }
    __threadfence();
  }
  __syncthreads();
}

// ---------------------------------------------------------------------------
// Weight packing: W (N x K row-major f32) -> B-fragment tiles f16.
// ---------------------------------------------------------------------------
__global__ void pack_b_kernel(const float* __restrict__ W, half_t* __restrict__ out,
                              int N, int K) {
  const int KT = K >> 5;
  const size_t total = (size_t)N * K;
  for (size_t i = (size_t)blockIdx.x * blockDim.x + threadIdx.x; i < total;
       i += (size_t)gridDim.x * blockDim.x) {
    int h = (int)(i & 15);
    int lane = (int)((i >> 4) & 31);
    size_t rest = i >> 9;
    int kt = (int)(rest % KT);
    int nt = (int)(rest / KT);
    int n = nt * 16 + (lane & 15);
    int k = kt * 32 + ((lane >> 4) << 3) + (h & 7) + ((h >> 3) << 4);
    out[i] = (half_t)W[(size_t)n * K + k];
  }
}

// ---------------------------------------------------------------------------
// Encoder layer (bidirectional, persistent), templated on LAYER so the A-source
// select is compile-time. 8 WGs: (dir 0..1) x (seg 0..3); each WG owns hidden
// cols [seg*128, seg*128+128) of its direction.
// ---------------------------------------------------------------------------
template <int LAYER>
__global__ void __launch_bounds__(256, 2) enc_layer_kernel(
    const float* __restrict__ x,      // (B,T,64)   layer 0 input
    const half_t* __restrict__ xs1,   // (T*B,1024) layer 1 input
    const half_t* __restrict__ pWi,   // [dir][96][KTI][512]
    const half_t* __restrict__ pWh,   // [dir][96][16][512]
    const float* __restrict__ bi, const float* __restrict__ bh,  // [dir][1536]
    float* __restrict__ h32, half_t* __restrict__ h16,           // [dir][128*512]
    half_t* __restrict__ yout,        // (T*B,1024): layer0 writes dir*512+col
    int* bar_cnt, int* bar_gen) {
  extern __shared__ half_t smem[];
  constexpr int KTI = (LAYER == 0) ? 2 : 32;
  const int tid = threadIdx.x, wave = tid >> 5, lane = tid & 31;
  const int wg = blockIdx.x, dir = wg >> 2, seg = wg & 3;
  const half_t* Wi = pWi + (size_t)dir * 96 * KTI * 512;
  const half_t* Wh = pWh + (size_t)dir * 96 * 16 * 512;
  const float* biD = bi + dir * NG;
  const float* bhD = bh + dir * NG;
  float* h32d = h32 + (size_t)dir * B_ * H_;
  half_t* h16d = h16 + (size_t)dir * B_ * H_;
  const u32 lds_off = (u32)(uintptr_t)(void*)smem;

  for (int i = tid; i < B_ * H_; i += 256) { h32d[i] = 0.0f; h16d[i] = (half_t)0.0f; }
  grid_barrier(bar_cnt, bar_gen, 8);

  for (int step = 0; step < T_; ++step) {
    const int t = dir ? (T_ - 1 - step) : step;
    if (wave == 0) {                       // DMA full h (f16) into LDS
      tdm_load_tile(lds_off, h16d, B_);
      __builtin_amdgcn_s_wait_tensorcnt(0);
    }
    __syncthreads();
    grid_barrier(bar_cnt, bar_gen, 8);     // all TDM reads done before h writes

    for (int job = wave; job < 64; job += 8) {
      const int mt = job >> 3, jt = job & 7, m0 = mt * 16;
      const int hc = seg * 128 + jt * 16 + (lane & 15);
      v8f aI[3], aH[3];
#pragma unroll
      for (int g = 0; g < 3; g++) { aI[g] = zero8(); aH[g] = zero8(); }
      // input projection gi = x_t @ Wi^T
      for (int kt = 0; kt < KTI; ++kt) {
        const int row = m0 + (lane & 15);
        const int klo = kt * 32 + ((lane >> 4) << 3);
        v16h a;
        if (LAYER == 0)
          a = ld_a_f32(x + (size_t)row * T_ * 64 + (size_t)t * 64 + klo);
        else
          a = ld_a_f16g(xs1 + ((size_t)t * B_ + row) * 1024 + klo);
#pragma unroll
        for (int g = 0; g < 3; g++)
          aI[g] = wmma16(a, ld_b_frag(Wi, g * 32 + seg * 8 + jt, kt, KTI), aI[g]);
      }
      // recurrent gh = h @ Wh^T (A from LDS)
      for (int kt = 0; kt < 16; ++kt) {
        v16h a = ld_a_lds(smem, m0, kt * 32);
#pragma unroll
        for (int g = 0; g < 3; g++)
          aH[g] = wmma16(a, ld_b_frag(Wh, g * 32 + seg * 8 + jt, kt, 16), aH[g]);
      }
      // gate combine (C-fragment layout)
      const float br = biD[hc], bz = biD[512 + hc], bn = biD[1024 + hc];
      const float cr = bhD[hc], cz = bhD[512 + hc], cn = bhD[1024 + hc];
#pragma unroll
      for (int v = 0; v < 8; v++) {
        const int row = m0 + v + ((lane >> 4) << 3);
        const float hp = h32d[(size_t)row * H_ + hc];
        const float r = sigm(aI[0][v] + br + aH[0][v] + cr);
        const float z = sigm(aI[1][v] + bz + aH[1][v] + cz);
        const float n = tanhf(aI[2][v] + bn + r * (aH[2][v] + cn));
        const float hn = (1.0f - z) * n + z * hp;
        h32d[(size_t)row * H_ + hc] = hn;
        h16d[(size_t)row * H_ + hc] = (half_t)hn;
        if (LAYER == 0)
          yout[((size_t)t * B_ + row) * 1024 + dir * 512 + hc] = (half_t)hn;
      }
    }
    grid_barrier(bar_cnt, bar_gen, 8);
  }
}

// ---------------------------------------------------------------------------
// Encoder linear: state = [hb1, hf1] @ enc_lin_W^T + b -> decoder init h0,h1.
// K-loop split in two straight halves (hb then hf), no per-iteration branch.
// ---------------------------------------------------------------------------
__global__ void __launch_bounds__(256) enc_lin_kernel(
    const float* __restrict__ hf, const float* __restrict__ hb,
    const half_t* __restrict__ pLin, const float* __restrict__ b,
    float* __restrict__ h0_32, half_t* __restrict__ h0_16,
    float* __restrict__ h1_32, half_t* __restrict__ h1_16) {
  const int id = blockIdx.x * 8 + (threadIdx.x >> 5);
  const int lane = threadIdx.x & 31;
  const int nt = id & 63, mt = id >> 6;
  const int m0 = mt * 16, n0 = nt * 16;
  const int row = m0 + (lane & 15);
  v8f acc = zero8();
  for (int kt = 0; kt < 16; ++kt) {      // enc cols 0..511 = hb1
    const int klo = kt * 32 + ((lane >> 4) << 3);
    acc = wmma16(ld_a_f32(hb + (size_t)row * 512 + klo),
                 ld_b_frag(pLin, nt, kt, 32), acc);
  }
  for (int kt = 16; kt < 32; ++kt) {     // enc cols 512..1023 = hf1
    const int klo = (kt - 16) * 32 + ((lane >> 4) << 3);
    acc = wmma16(ld_a_f32(hf + (size_t)row * 512 + klo),
                 ld_b_frag(pLin, nt, kt, 32), acc);
  }
  const int col = n0 + (lane & 15);
  const float bias = b[col];
  float* H32 = (col < 512) ? h0_32 : h1_32;
  half_t* H16 = (col < 512) ? h0_16 : h1_16;
  const int c = col & 511;
#pragma unroll
  for (int v = 0; v < 8; v++) {
    const int rr = m0 + v + ((lane >> 4) << 3);
    const float val = acc[v] + bias;
    H32[(size_t)rr * 512 + c] = val;
    H16[(size_t)rr * 512 + c] = (half_t)val;
  }
}

// ---------------------------------------------------------------------------
// Decoder (persistent). 4 WGs (seg 0..3 own 128-wide hidden slices).
// Per step: cell0, cell1 (input+hidden GEMMs), output proj + log_softmax.
// ---------------------------------------------------------------------------
__global__ void __launch_bounds__(256, 2) decoder_kernel(
    const float* __restrict__ y,  // (B,U,64)
    const half_t* __restrict__ pWi0, const half_t* __restrict__ pWh0,
    const half_t* __restrict__ pWi1, const half_t* __restrict__ pWh1,
    const half_t* __restrict__ pOutW,
    const float* __restrict__ bi0, const float* __restrict__ bh0,
    const float* __restrict__ bi1, const float* __restrict__ bh1,
    const float* __restrict__ ob,
    float* __restrict__ hd0_32, half_t* __restrict__ hd0_16,
    float* __restrict__ hd1_32, half_t* __restrict__ hd1_16,
    float* __restrict__ logp, int* bar_cnt, int* bar_gen) {
  extern __shared__ half_t smem[];
  half_t* L0 = smem;
  half_t* L1 = smem + (size_t)B_ * LDH;
  const int tid = threadIdx.x, wave = tid >> 5, lane = tid & 31;
  const int seg = blockIdx.x;
  const u32 offL0 = (u32)(uintptr_t)(void*)L0;
  const u32 offL1 = (u32)(uintptr_t)(void*)L1;
  float lacc[8];
#pragma unroll
  for (int v = 0; v < 8; v++) lacc[v] = 0.0f;

  for (int t = 0; t < U_; ++t) {
    // ---- stage h0 ----
    if (wave == 0) { tdm_load_tile(offL0, hd0_16, B_); __builtin_amdgcn_s_wait_tensorcnt(0); }
    __syncthreads();
    grid_barrier(bar_cnt, bar_gen, 4);
    // ---- cell 0 ----
    for (int job = wave; job < 64; job += 8) {
      const int mt = job >> 3, jt = job & 7, m0 = mt * 16;
      const int hc = seg * 128 + jt * 16 + (lane & 15);
      v8f aI[3], aH[3];
#pragma unroll
      for (int g = 0; g < 3; g++) { aI[g] = zero8(); aH[g] = zero8(); }
      if (t > 0) {    // t==0 -> y_in = 0 -> input GEMM contributes nothing
        const int row = m0 + (lane & 15);
        const float* yb = y + (size_t)row * U_ * 64 + (size_t)(t - 1) * 64;
#pragma unroll
        for (int kt = 0; kt < 2; ++kt) {
          v16h a = ld_a_f32(yb + kt * 32 + ((lane >> 4) << 3));
#pragma unroll
          for (int g = 0; g < 3; g++)
            aI[g] = wmma16(a, ld_b_frag(pWi0, g * 32 + seg * 8 + jt, kt, 2), aI[g]);
        }
      }
      for (int kt = 0; kt < 16; ++kt) {
        v16h a = ld_a_lds(L0, m0, kt * 32);
#pragma unroll
        for (int g = 0; g < 3; g++)
          aH[g] = wmma16(a, ld_b_frag(pWh0, g * 32 + seg * 8 + jt, kt, 16), aH[g]);
      }
      const float br = bi0[hc], bz = bi0[512 + hc], bn = bi0[1024 + hc];
      const float cr = bh0[hc], cz = bh0[512 + hc], cn = bh0[1024 + hc];
#pragma unroll
      for (int v = 0; v < 8; v++) {
        const int row = m0 + v + ((lane >> 4) << 3);
        const float hp = hd0_32[(size_t)row * H_ + hc];
        const float r = sigm(aI[0][v] + br + aH[0][v] + cr);
        const float z = sigm(aI[1][v] + bz + aH[1][v] + cz);
        const float n = tanhf(aI[2][v] + bn + r * (aH[2][v] + cn));
        const float hn = (1.0f - z) * n + z * hp;
        hd0_32[(size_t)row * H_ + hc] = hn;
        hd0_16[(size_t)row * H_ + hc] = (half_t)hn;
      }
    }
    grid_barrier(bar_cnt, bar_gen, 4);
    // ---- stage h0', h1 ----
    if (wave == 0) {
      tdm_load_tile(offL0, hd0_16, B_);
      tdm_load_tile(offL1, hd1_16, B_);
      __builtin_amdgcn_s_wait_tensorcnt(0);
    }
    __syncthreads();
    grid_barrier(bar_cnt, bar_gen, 4);
    // ---- cell 1 (two separate k-loops to halve live B-fragment streams) ----
    for (int job = wave; job < 64; job += 8) {
      const int mt = job >> 3, jt = job & 7, m0 = mt * 16;
      const int hc = seg * 128 + jt * 16 + (lane & 15);
      v8f aI[3], aH[3];
#pragma unroll
      for (int g = 0; g < 3; g++) { aI[g] = zero8(); aH[g] = zero8(); }
      for (int kt = 0; kt < 16; ++kt) {
        v16h a0 = ld_a_lds(L0, m0, kt * 32);
#pragma unroll
        for (int g = 0; g < 3; g++)
          aI[g] = wmma16(a0, ld_b_frag(pWi1, g * 32 + seg * 8 + jt, kt, 16), aI[g]);
      }
      for (int kt = 0; kt < 16; ++kt) {
        v16h a1 = ld_a_lds(L1, m0, kt * 32);
#pragma unroll
        for (int g = 0; g < 3; g++)
          aH[g] = wmma16(a1, ld_b_frag(pWh1, g * 32 + seg * 8 + jt, kt, 16), aH[g]);
      }
      const float br = bi1[hc], bz = bi1[512 + hc], bn = bi1[1024 + hc];
      const float cr = bh1[hc], cz = bh1[512 + hc], cn = bh1[1024 + hc];
#pragma unroll
      for (int v = 0; v < 8; v++) {
        const int row = m0 + v + ((lane >> 4) << 3);
        const float hp = hd1_32[(size_t)row * H_ + hc];
        const float r = sigm(aI[0][v] + br + aH[0][v] + cr);
        const float z = sigm(aI[1][v] + bz + aH[1][v] + cz);
        const float n = tanhf(aI[2][v] + bn + r * (aH[2][v] + cn));
        const float hn = (1.0f - z) * n + z * hp;
        hd1_32[(size_t)row * H_ + hc] = hn;
        hd1_16[(size_t)row * H_ + hc] = (half_t)hn;
      }
    }
    grid_barrier(bar_cnt, bar_gen, 4);
    // ---- stage h1' ----
    if (wave == 0) { tdm_load_tile(offL1, hd1_16, B_); __builtin_amdgcn_s_wait_tensorcnt(0); }
    __syncthreads();
    // ---- output projection + log_softmax + score (waves 0,1 only) ----
    if (wave < 2) {
      const int m0 = (seg * 2 + wave) * 16;
      v8f acc[4];
#pragma unroll
      for (int nt = 0; nt < 4; nt++) acc[nt] = zero8();
      for (int kt = 0; kt < 16; ++kt) {
        v16h a = ld_a_lds(L1, m0, kt * 32);
#pragma unroll
        for (int nt = 0; nt < 4; nt++)
          acc[nt] = wmma16(a, ld_b_frag(pOutW, nt, kt, 16), acc[nt]);
      }
      const float b0 = ob[(lane & 15)], b1 = ob[16 + (lane & 15)];
      const float b2 = ob[32 + (lane & 15)], b3 = ob[48 + (lane & 15)];
#pragma unroll
      for (int v = 0; v < 8; v++) {
        float a0 = acc[0][v] + b0, a1 = acc[1][v] + b1;
        float a2 = acc[2][v] + b2, a3 = acc[3][v] + b3;
        float mx = fmaxf(fmaxf(a0, a1), fmaxf(a2, a3));
#pragma unroll
        for (int m = 1; m < 16; m <<= 1) mx = fmaxf(mx, __shfl_xor(mx, m));
        float s = expf(a0 - mx) + expf(a1 - mx) + expf(a2 - mx) + expf(a3 - mx);
#pragma unroll
        for (int m = 1; m < 16; m <<= 1) s += __shfl_xor(s, m);
        const float ls = mx + logf(s);
        const int row = m0 + v + ((lane >> 4) << 3);
        const float* yr = y + (size_t)row * U_ * 64 + (size_t)t * 64 + (lane & 15);
        lacc[v] += (a0 - ls) * yr[0] + (a1 - ls) * yr[16] + (a2 - ls) * yr[32] +
                   (a3 - ls) * yr[48];
      }
    }
    grid_barrier(bar_cnt, bar_gen, 4);
  }
  if (wave < 2) {
    const int m0 = (seg * 2 + wave) * 16;
#pragma unroll
    for (int v = 0; v < 8; v++) {
      float s = lacc[v];
#pragma unroll
      for (int m = 1; m < 16; m <<= 1) s += __shfl_xor(s, m);
      if ((lane & 15) == 0) logp[m0 + v + ((lane >> 4) << 3)] = s;
    }
  }
}

// ---------------------------------------------------------------------------
// Host launcher
// ---------------------------------------------------------------------------
extern "C" void kernel_launch(void* const* d_in, const int* in_sizes, int n_in,
                              void* d_out, int out_size, void* d_ws, size_t ws_size,
                              hipStream_t stream) {
  (void)in_sizes; (void)n_in; (void)out_size; (void)ws_size;
  const float* x      = (const float*)d_in[0];
  const float* y      = (const float*)d_in[1];
  const float* eWi0   = (const float*)d_in[2];
  const float* eWh0   = (const float*)d_in[3];
  const float* ebi0   = (const float*)d_in[4];
  const float* ebh0   = (const float*)d_in[5];
  const float* eWi1   = (const float*)d_in[6];
  const float* eWh1   = (const float*)d_in[7];
  const float* ebi1   = (const float*)d_in[8];
  const float* ebh1   = (const float*)d_in[9];
  const float* eLinW  = (const float*)d_in[10];
  const float* eLinb  = (const float*)d_in[11];
  const float* dWi0   = (const float*)d_in[12];
  const float* dWh0   = (const float*)d_in[13];
  const float* dbi0   = (const float*)d_in[14];
  const float* dbh0   = (const float*)d_in[15];
  const float* dWi1   = (const float*)d_in[16];
  const float* dWh1   = (const float*)d_in[17];
  const float* dbi1   = (const float*)d_in[18];
  const float* dbh1   = (const float*)d_in[19];
  const float* oW     = (const float*)d_in[20];
  const float* ob     = (const float*)d_in[21];

  size_t off = 0;
  auto take = [&](size_t bytes) -> void* {
    void* p = (char*)d_ws + off;
    off = (off + bytes + 255) & ~(size_t)255;
    return p;
  };
  half_t* pWi0  = (half_t*)take((size_t)3072 * 64 * 2);
  half_t* pWh0  = (half_t*)take((size_t)3072 * 512 * 2);
  half_t* pWi1  = (half_t*)take((size_t)3072 * 1024 * 2);
  half_t* pWh1  = (half_t*)take((size_t)3072 * 512 * 2);
  half_t* pLin  = (half_t*)take((size_t)1024 * 1024 * 2);
  half_t* pDWi0 = (half_t*)take((size_t)1536 * 64 * 2);
  half_t* pDWh0 = (half_t*)take((size_t)1536 * 512 * 2);
  half_t* pDWi1 = (half_t*)take((size_t)1536 * 512 * 2);
  half_t* pDWh1 = (half_t*)take((size_t)1536 * 512 * 2);
  half_t* pOutW = (half_t*)take((size_t)64 * 512 * 2);
  half_t* xs1   = (half_t*)take((size_t)T_ * B_ * 1024 * 2);
  float*  hL0_32 = (float*)take((size_t)2 * B_ * H_ * 4);
  half_t* hL0_16 = (half_t*)take((size_t)2 * B_ * H_ * 2);
  float*  hL1_32 = (float*)take((size_t)2 * B_ * H_ * 4);
  half_t* hL1_16 = (half_t*)take((size_t)2 * B_ * H_ * 2);
  float*  hd0_32 = (float*)take((size_t)B_ * H_ * 4);
  half_t* hd0_16 = (half_t*)take((size_t)B_ * H_ * 2);
  float*  hd1_32 = (float*)take((size_t)B_ * H_ * 4);
  half_t* hd1_16 = (half_t*)take((size_t)B_ * H_ * 2);
  int*    bars   = (int*)take(256);

  hipMemsetAsync(bars, 0, 256, stream);

  auto pack = [&](const float* W, half_t* out, int N, int K) {
    size_t tot = (size_t)N * K;
    int blocks = (int)((tot + 255) / 256);
    if (blocks > 4096) blocks = 4096;
    pack_b_kernel<<<blocks, 256, 0, stream>>>(W, out, N, K);
  };
  pack(eWi0, pWi0, 3072, 64);
  pack(eWh0, pWh0, 3072, 512);
  pack(eWi1, pWi1, 3072, 1024);
  pack(eWh1, pWh1, 3072, 512);
  pack(eLinW, pLin, 1024, 1024);
  pack(dWi0, pDWi0, 1536, 64);
  pack(dWh0, pDWh0, 1536, 512);
  pack(dWi1, pDWi1, 1536, 512);
  pack(dWh1, pDWh1, 1536, 512);
  pack(oW, pOutW, 64, 512);

  const size_t encLds = (size_t)B_ * LDH * 2;       // 133,120 B
  const size_t decLds = 2 * encLds;                 // 266,240 B
  hipFuncSetAttribute(reinterpret_cast<const void*>(&enc_layer_kernel<0>),
                      hipFuncAttributeMaxDynamicSharedMemorySize, (int)encLds);
  hipFuncSetAttribute(reinterpret_cast<const void*>(&enc_layer_kernel<1>),
                      hipFuncAttributeMaxDynamicSharedMemorySize, (int)encLds);
  hipFuncSetAttribute(reinterpret_cast<const void*>(&decoder_kernel),
                      hipFuncAttributeMaxDynamicSharedMemorySize, (int)decLds);

  enc_layer_kernel<0><<<8, 256, encLds, stream>>>(
      x, (const half_t*)nullptr, pWi0, pWh0, ebi0, ebh0,
      hL0_32, hL0_16, xs1, bars + 0, bars + 1);
  enc_layer_kernel<1><<<8, 256, encLds, stream>>>(
      x, xs1, pWi1, pWh1, ebi1, ebh1,
      hL1_32, hL1_16, (half_t*)nullptr, bars + 16, bars + 17);
  // enc = [hb1 (cols 0..511), hf1 (cols 512..1023)]
  enc_lin_kernel<<<64, 256, 0, stream>>>(
      hL1_32 /*fwd*/, hL1_32 + (size_t)B_ * H_ /*bwd*/, pLin, eLinb,
      hd0_32, hd0_16, hd1_32, hd1_16);
  decoder_kernel<<<4, 256, decLds, stream>>>(
      y, pDWi0, pDWh0, pDWi1, pDWh1, pOutW,
      dbi0, dbh0, dbi1, dbh1, ob,
      hd0_32, hd0_16, hd1_32, hd1_16,
      (float*)d_out, bars + 32, bars + 33);
}
